// ObjectMoELayer_10651518894850
// MI455X (gfx1250) — compile-verified, matrix operand order checked by reading, not run on hardware
//
#include <hip/hip_runtime.h>
#include <hip/hip_bf16.h>
#include <cstdint>

typedef __attribute__((ext_vector_type(16))) __bf16 v16bf;
typedef __attribute__((ext_vector_type(8)))  __bf16 v8bf;
typedef __attribute__((ext_vector_type(8)))  float  v8f;
typedef __attribute__((ext_vector_type(4)))  int    v4i;

namespace {
constexpr int kBsz = 8, kNOBJ = 16, kT = 64, kD = 1024, kDO = 1024;
constexpr int kE = 8, kES = 2, kEH = 2048, kH = 8, kFH = 4096;
constexpr int kN  = kBsz * kNOBJ * kT;   // 8192 tokens
constexpr int kS  = kNOBJ * kT;          // 1024 seq per batch
constexpr int kDH = kDO / kH;            // 128 head dim
}

enum { GF_RELU = 1, GF_ACCUM = 2, GF_TRANSB = 4 };

// ---- CDNA5 async global->LDS copy (ASYNCcnt-tracked), guarded fallback ----
#if defined(__has_builtin)
#  if __has_builtin(__builtin_amdgcn_global_load_async_to_lds_b128)
#    define HAVE_ASYNC_LDS 1
#  endif
#endif
#ifndef HAVE_ASYNC_LDS
#  define HAVE_ASYNC_LDS 0
#endif

typedef __attribute__((address_space(1))) v4i* gas_v4i_p;
typedef __attribute__((address_space(3))) v4i* las_v4i_p;

__device__ __forceinline__ void stage16(const __bf16* g, __bf16* l) {
#if HAVE_ASYNC_LDS
  __builtin_amdgcn_global_load_async_to_lds_b128(
      (gas_v4i_p)(void*)g, (las_v4i_p)(void*)l, 0, 0);
#else
  *reinterpret_cast<v8bf*>(l) = *reinterpret_cast<const v8bf*>(g);
#endif
}

__device__ __forceinline__ void stage_wait() {
#if HAVE_ASYNC_LDS
#  if __has_builtin(__builtin_amdgcn_s_wait_asynccnt)
  __builtin_amdgcn_s_wait_asynccnt(0);
#  else
  asm volatile("s_wait_asynccnt 0x0" ::: "memory");
#  endif
#endif
}

__device__ __forceinline__ v8f zero_v8f() {
  v8f z;
#pragma unroll
  for (int i = 0; i < 8; ++i) z[i] = 0.0f;
  return z;
}

// One 16x16x32 bf16 WMMA fragment half-row: 8 contiguous bf16 at kbase plus
// 8 contiguous bf16 at kbase+16 (ISA 16-bit A/B-matrix layout, wave32).
__device__ __forceinline__ v16bf load_frag16(const __bf16* p) {
  v8bf lo = *reinterpret_cast<const v8bf*>(p);
  v8bf hi = *reinterpret_cast<const v8bf*>(p + 16);
  v16bf f;
#pragma unroll
  for (int i = 0; i < 8; ++i) { f[i] = lo[i]; f[i + 8] = hi[i]; }
  return f;
}

// ---------------------------------------------------------------------------
// fp32 -> bf16 (plain), one HBM pass
// ---------------------------------------------------------------------------
__global__ void cvt_f32_bf16(const float* __restrict__ in,
                             __bf16* __restrict__ out, size_t n) {
  size_t stride = (size_t)gridDim.x * blockDim.x;
  for (size_t i = (size_t)blockIdx.x * blockDim.x + threadIdx.x; i < n; i += stride)
    out[i] = static_cast<__bf16>(in[i]);
}

// ---------------------------------------------------------------------------
// fp32 (K x N, row-major, batched) -> bf16 transposed (N x K): weights become
// Bt so GEMM staging is contiguous on both operands (no in-loop transpose).
// block (32,8); tile 32x32 through LDS; both global accesses coalesced.
// ---------------------------------------------------------------------------
__global__ __launch_bounds__(256) void cvt_t_f32_bf16(
    const float* __restrict__ in, __bf16* __restrict__ out, int K, int N) {
  __shared__ float tile[32][33];
  const size_t base_in  = (size_t)blockIdx.z * K * N;
  const size_t base_out = (size_t)blockIdx.z * K * N;
  int k0 = blockIdx.y * 32, n0 = blockIdx.x * 32;
  int tx = threadIdx.x, ty = threadIdx.y;
#pragma unroll
  for (int i = 0; i < 32; i += 8)
    tile[ty + i][tx] = in[base_in + (size_t)(k0 + ty + i) * N + n0 + tx];
  __syncthreads();
#pragma unroll
  for (int i = 0; i < 32; i += 8)
    out[base_out + (size_t)(n0 + ty + i) * K + k0 + tx] =
        static_cast<__bf16>(tile[tx][ty + i]);
}

// ---------------------------------------------------------------------------
// Router: logits = xf @ Wg (N x 8), top-2 softmax -> dense gates (N x 8)
// ---------------------------------------------------------------------------
__global__ void router_kernel(const float* __restrict__ x,
                              const float* __restrict__ Wg,
                              float* __restrict__ gates) {
  int tok = blockIdx.x * blockDim.x + threadIdx.x;
  if (tok >= kN) return;
  float acc[kE];
#pragma unroll
  for (int e = 0; e < kE; ++e) acc[e] = 0.0f;
  const float* xr = x + (size_t)tok * kD;
  for (int d = 0; d < kD; ++d) {
    float xv = xr[d];
    const float* wr = Wg + (size_t)d * kE;
#pragma unroll
    for (int e = 0; e < kE; ++e) acc[e] += xv * wr[e];
  }
  int i0 = 0;
#pragma unroll
  for (int e = 1; e < kE; ++e) if (acc[e] > acc[i0]) i0 = e;
  int i1 = (i0 == 0) ? 1 : 0;
#pragma unroll
  for (int e = 0; e < kE; ++e) if (e != i0 && acc[e] > acc[i1]) i1 = e;
  float ex = __expf(acc[i1] - acc[i0]);
  float g0 = 1.0f / (1.0f + ex);
  float g1 = ex / (1.0f + ex);
  float* g = gates + (size_t)tok * kE;
#pragma unroll
  for (int e = 0; e < kE; ++e) g[e] = 0.0f;
  g[i0] = g0;
  g[i1] = g1;
}

// ---------------------------------------------------------------------------
// bf16 WMMA GEMM: C[MxN] = A[MxK] @ Bt[NxK]^T (+bias/relu/rowscale/accum,
// fp32 and/or bf16 store, optional transposed bf16 store).
// 256 threads = 8 waves; block tile 128x128; wave tile 32x64; K staged 32
// wide, double-buffered LDS, async global->LDS pipeline (1 barrier / step).
// ---------------------------------------------------------------------------
__global__ __launch_bounds__(256) void gemm_bf16(
    const __bf16* __restrict__ A, const __bf16* __restrict__ Bt,
    const float* __restrict__ bias, const float* __restrict__ rowscale,
    int rs_stride, float* __restrict__ outF, __bf16* __restrict__ outB,
    int M, int N, int Kd, int flags) {
  constexpr int LS = 40;                       // 32 + 8 pad
  __shared__ __bf16 lA[2][128 * LS];           // [m][k]
  __shared__ __bf16 lB[2][128 * LS];           // [n][k]

  const int tid  = threadIdx.x;
  const int wave = tid >> 5, lane = tid & 31;
  const int hi   = lane >> 4, ln = lane & 15;
  const int bm   = blockIdx.y * 128;
  const int bn   = blockIdx.x * 128;
  const int wm   = (wave >> 1) * 32;
  const int wn   = (wave & 1) * 64;

  v8f acc[2][4];
#pragma unroll
  for (int i = 0; i < 2; ++i)
#pragma unroll
    for (int j = 0; j < 4; ++j) acc[i][j] = zero_v8f();

  // staging map: each thread copies 16 elems of A and 16 of Bt per step
  const int sr = tid >> 1, sc = (tid & 1) * 16;
  const __bf16* gA = A  + (size_t)(bm + sr) * Kd + sc;
  const __bf16* gB = Bt + (size_t)(bn + sr) * Kd + sc;

  auto stage = [&](int k0, int buf) {
    stage16(gA + k0,     lA[buf] + sr * LS + sc);
    stage16(gA + k0 + 8, lA[buf] + sr * LS + sc + 8);
    stage16(gB + k0,     lB[buf] + sr * LS + sc);
    stage16(gB + k0 + 8, lB[buf] + sr * LS + sc + 8);
  };

  const int nk = Kd >> 5;
  stage(0, 0);
  const int kb = hi * 8;
  for (int t = 0; t < nk; ++t) {
    stage_wait();
    __syncthreads();
    if (t + 1 < nk) stage((t + 1) << 5, (t + 1) & 1);

    const __bf16* bufA = lA[t & 1];
    const __bf16* bufB = lB[t & 1];
    v16bf af[2], bfr[4];
#pragma unroll
    for (int i = 0; i < 2; ++i)
      af[i] = load_frag16(bufA + (wm + i * 16 + ln) * LS + kb);
#pragma unroll
    for (int j = 0; j < 4; ++j)
      bfr[j] = load_frag16(bufB + (wn + j * 16 + ln) * LS + kb);
#pragma unroll
    for (int i = 0; i < 2; ++i)
#pragma unroll
      for (int j = 0; j < 4; ++j)
        acc[i][j] = __builtin_amdgcn_wmma_f32_16x16x32_bf16(
            false, af[i], false, bfr[j], (short)0, acc[i][j], false, false);
  }

  // epilogue (D layout: m = v + hi*8, n = lane&15)
#pragma unroll
  for (int i = 0; i < 2; ++i)
#pragma unroll
    for (int j = 0; j < 4; ++j)
#pragma unroll
      for (int v = 0; v < 8; ++v) {
        float val = acc[i][j][v];
        int gm = bm + wm + i * 16 + v + hi * 8;
        int gn = bn + wn + j * 16 + ln;
        if (bias) val += bias[gn];
        if (flags & GF_RELU) val = fmaxf(val, 0.0f);
        if (rowscale) val *= rowscale[(size_t)gm * rs_stride];
        size_t idx = (size_t)gm * N + gn;
        if (outF) {
          if (flags & GF_ACCUM) outF[idx] += val;
          else                  outF[idx]  = val;
        }
        if (outB) {
          size_t ob = (flags & GF_TRANSB) ? ((size_t)gn * M + gm) : idx;
          outB[ob] = static_cast<__bf16>(val);
        }
      }
}

// ---------------------------------------------------------------------------
// Flash attention: one (b,h) + 64 queries per block (16/wave), 32-key chunks,
// double-buffered async K/V staging. V arrives pre-transposed ([dh][token])
// so both K and Vt staging are contiguous 16B copies.
// ---------------------------------------------------------------------------
__global__ __launch_bounds__(128) void attn_kernel(
    const __bf16* __restrict__ Q, const __bf16* __restrict__ Kmat,
    const __bf16* __restrict__ Vt, __bf16* __restrict__ ctx) {
  constexpr int KLS = 136;   // 128 + 8 pad
  constexpr int PLS = 40;    // 32 + 8 pad
  __shared__ __bf16 lK[2][32 * KLS];     // [key][dh]
  __shared__ __bf16 lVt[2][128 * PLS];   // [dh][key]
  __shared__ __bf16 lP[4 * 16 * PLS];    // per-wave P tile [q][key]

  const int bh = blockIdx.y;
  const int b = bh / kH, h = bh % kH;
  const int wave = threadIdx.x >> 5, lane = threadIdx.x & 31;
  const int hi = lane >> 4, ln = lane & 15;
  const int qbase = blockIdx.x * 64 + wave * 16;
  __bf16* myP = lP + wave * 16 * PLS;
  const float sm_scale = 0.08838834764831845f;   // 1/sqrt(128)

  // Q fragments (16 queries x 128 dh), loaded once
  const size_t qoff = ((size_t)b * kS + qbase + ln) * kDO + h * kDH;
  v16bf qf[4];
#pragma unroll
  for (int c = 0; c < 4; ++c)
    qf[c] = load_frag16(Q + qoff + c * 32 + hi * 8);

  // staging map (128 threads)
  const int skey = threadIdx.x >> 2;            // 0..31
  const int sdb  = (threadIdx.x & 3) * 32;      // dh chunk
  const __bf16* gK = Kmat + ((size_t)b * kS + skey) * kDO + h * kDH + sdb;
  const __bf16* gV = Vt + ((size_t)(h * kDH + threadIdx.x)) * kN + (size_t)b * kS;

  auto stage = [&](int kc, int buf) {
#pragma unroll
    for (int i = 0; i < 4; ++i)
      stage16(gK + (size_t)kc * kDO + i * 8,
              lK[buf] + skey * KLS + sdb + i * 8);
#pragma unroll
    for (int i = 0; i < 4; ++i)
      stage16(gV + kc + i * 8, lVt[buf] + threadIdx.x * PLS + i * 8);
  };

  float mrow[8], lrow[8];
#pragma unroll
  for (int v = 0; v < 8; ++v) { mrow[v] = -1e30f; lrow[v] = 0.0f; }
  v8f o[8];
#pragma unroll
  for (int j = 0; j < 8; ++j) o[j] = zero_v8f();

  const int nc = kS >> 5;
  stage(0, 0);
  for (int t = 0; t < nc; ++t) {
    stage_wait();
    __syncthreads();
    if (t + 1 < nc) stage((t + 1) << 5, (t + 1) & 1);

    const __bf16* bK = lK[t & 1];
    const __bf16* bV = lVt[t & 1];

    // scores: 16 queries x 32 keys, K-dim = dh (128)
    v8f s[2];
#pragma unroll
    for (int nt = 0; nt < 2; ++nt) {
      v8f a = zero_v8f();
#pragma unroll
      for (int c = 0; c < 4; ++c) {
        v16bf kf = load_frag16(bK + (nt * 16 + ln) * KLS + c * 32 + hi * 8);
        a = __builtin_amdgcn_wmma_f32_16x16x32_bf16(
            false, qf[c], false, kf, (short)0, a, false, false);
      }
      s[nt] = a;
    }

    // online softmax per row v (row m = v + hi*8; 16-lane butterfly)
    float scalef[8];
#pragma unroll
    for (int v = 0; v < 8; ++v) {
      float s0 = s[0][v] * sm_scale;
      float s1 = s[1][v] * sm_scale;
      float mx = fmaxf(s0, s1);
#pragma unroll
      for (int off = 1; off < 16; off <<= 1)
        mx = fmaxf(mx, __shfl_xor(mx, off, 32));
      float mnew = fmaxf(mrow[v], mx);
      float sc = __expf(mrow[v] - mnew);
      float p0 = __expf(s0 - mnew);
      float p1 = __expf(s1 - mnew);
      float ps = p0 + p1;
#pragma unroll
      for (int off = 1; off < 16; off <<= 1)
        ps += __shfl_xor(ps, off, 32);
      lrow[v] = lrow[v] * sc + ps;
      mrow[v] = mnew;
      scalef[v] = sc;
      myP[(v + hi * 8) * PLS + ln]      = static_cast<__bf16>(p0);
      myP[(v + hi * 8) * PLS + 16 + ln] = static_cast<__bf16>(p1);
    }
#pragma unroll
    for (int j = 0; j < 8; ++j)
#pragma unroll
      for (int v = 0; v < 8; ++v) o[j][v] *= scalef[v];

    // PV: A = P (16x32 keys), B = V chunk via [dh][key] LDS
    v16bf pf = load_frag16(myP + ln * PLS + hi * 8);
#pragma unroll
    for (int j = 0; j < 8; ++j) {
      v16bf vf = load_frag16(bV + (j * 16 + ln) * PLS + hi * 8);
      o[j] = __builtin_amdgcn_wmma_f32_16x16x32_bf16(
          false, pf, false, vf, (short)0, o[j], false, false);
    }
  }

  // finalize: divide by l, write bf16 ctx (token, H*dh layout)
#pragma unroll
  for (int j = 0; j < 8; ++j)
#pragma unroll
    for (int v = 0; v < 8; ++v) {
      float val = o[j][v] / lrow[v];
      int m = qbase + v + hi * 8;
      ctx[((size_t)b * kS + m) * kDO + h * kDH + j * 16 + ln] =
          static_cast<__bf16>(val);
    }
}

// ---------------------------------------------------------------------------
// Host-side orchestration
// ---------------------------------------------------------------------------
extern "C" void kernel_launch(void* const* d_in, const int* in_sizes, int n_in,
                              void* d_out, int out_size, void* d_ws, size_t ws_size,
                              hipStream_t stream) {
  (void)in_sizes; (void)n_in; (void)out_size; (void)ws_size;
  const float* x   = (const float*)d_in[0];
  const float* Wg  = (const float*)d_in[1];
  const float* We1 = (const float*)d_in[2];
  const float* be1 = (const float*)d_in[3];
  const float* We2 = (const float*)d_in[4];
  const float* be2 = (const float*)d_in[5];
  const float* Ws1 = (const float*)d_in[6];
  const float* bs1 = (const float*)d_in[7];
  const float* Ws2 = (const float*)d_in[8];
  const float* bs2 = (const float*)d_in[9];
  const float* Wq  = (const float*)d_in[10];
  const float* bq  = (const float*)d_in[11];
  const float* Wk  = (const float*)d_in[12];
  const float* bk  = (const float*)d_in[13];
  const float* Wv  = (const float*)d_in[14];
  const float* bv  = (const float*)d_in[15];
  const float* Wo  = (const float*)d_in[16];
  const float* bo  = (const float*)d_in[17];
  const float* Wf1 = (const float*)d_in[18];
  const float* bf1 = (const float*)d_in[19];
  const float* Wf2 = (const float*)d_in[20];
  const float* bf2 = (const float*)d_in[21];
  float* out = (float*)d_out;

  char* ws = (char*)d_ws;
  size_t off = 0;
  auto alloc = [&](size_t bytes) -> char* {
    char* p = ws + off;
    off += (bytes + 255) & ~(size_t)255;
    return p;
  };
  // All weight copies stored TRANSPOSED (Bt = N x K). Routed-expert weights
  // first: their 67MB region is dead after the MoE and aliases the FFN h1.
  __bf16* We1t = (__bf16*)alloc((size_t)kE * kD * kEH * 2);
  __bf16* We2t = (__bf16*)alloc((size_t)kE * kEH * kDO * 2);
  __bf16* Ws1t = (__bf16*)alloc((size_t)kES * kD * kEH * 2);
  __bf16* Ws2t = (__bf16*)alloc((size_t)kES * kEH * kDO * 2);
  __bf16* Wqt  = (__bf16*)alloc((size_t)kDO * kDO * 2);
  __bf16* Wkt  = (__bf16*)alloc((size_t)kDO * kDO * 2);
  __bf16* Wvt  = (__bf16*)alloc((size_t)kDO * kDO * 2);
  __bf16* Wot  = (__bf16*)alloc((size_t)kDO * kDO * 2);
  __bf16* Wf1t = (__bf16*)alloc((size_t)kDO * kFH * 2);
  __bf16* Wf2t = (__bf16*)alloc((size_t)kFH * kDO * 2);
  __bf16* xb   = (__bf16*)alloc((size_t)kN * kD * 2);
  float*  gates= (float*) alloc((size_t)kN * kE * 4);
  float*  moe  = (float*) alloc((size_t)kN * kDO * 4);
  __bf16* hbuf = (__bf16*)alloc((size_t)kN * kEH * 2);
  __bf16* objb = (__bf16*)alloc((size_t)kN * kDO * 2);
  __bf16* Qb   = (__bf16*)alloc((size_t)kN * kDO * 2);
  __bf16* Kb   = (__bf16*)alloc((size_t)kN * kDO * 2);
  __bf16* Vbt  = (__bf16*)alloc((size_t)kN * kDO * 2);  // transposed V
  __bf16* ctxb = (__bf16*)alloc((size_t)kN * kDO * 2);
  __bf16* relb = (__bf16*)alloc((size_t)kN * kDO * 2);
  __bf16* h1b  = We1t;   // alias: 8192*4096 bf16 = 64MiB <= We1t+We2t region

  auto cvt = [&](const float* src, __bf16* dst, size_t n) {
    int blocks = (int)((n + 255) / 256);
    if (blocks > 2048) blocks = 2048;
    cvt_f32_bf16<<<blocks, 256, 0, stream>>>(src, dst, n);
  };
  auto cvt_t = [&](const float* src, __bf16* dst, int K, int N, int batch) {
    cvt_t_f32_bf16<<<dim3(N / 32, K / 32, batch), dim3(32, 8), 0, stream>>>(
        src, dst, K, N);
  };
  auto gemm = [&](const __bf16* A, const __bf16* Bt, const float* bias,
                  const float* rowscale, int rs_stride, float* oF, __bf16* oB,
                  int M, int N, int Kd, int flags) {
    dim3 grid(N / 128, M / 128);
    gemm_bf16<<<grid, 256, 0, stream>>>(A, Bt, bias, rowscale, rs_stride,
                                        oF, oB, M, N, Kd, flags);
  };

  // ---- one HBM pass: fp32 -> bf16 (weights transposed; L2-resident after) --
  cvt(x, xb, (size_t)kN * kD);
  cvt_t(We1, We1t, kD, kEH, kE);
  cvt_t(We2, We2t, kEH, kDO, kE);
  cvt_t(Ws1, Ws1t, kD, kEH, kES);
  cvt_t(Ws2, Ws2t, kEH, kDO, kES);
  cvt_t(Wq, Wqt, kDO, kDO, 1);
  cvt_t(Wk, Wkt, kDO, kDO, 1);
  cvt_t(Wv, Wvt, kDO, kDO, 1);
  cvt_t(Wo, Wot, kDO, kDO, 1);
  cvt_t(Wf1, Wf1t, kDO, kFH, 1);
  cvt_t(Wf2, Wf2t, kFH, kDO, 1);

  // ---- router + gates ----
  router_kernel<<<kN / 256, 256, 0, stream>>>(x, Wg, gates);

  // ---- MoE: routed experts (dense, gate-masked like the reference) ----
  for (int e = 0; e < kE; ++e) {
    gemm(xb, We1t + (size_t)e * kD * kEH, be1 + (size_t)e * kEH,
         nullptr, 0, nullptr, hbuf, kN, kEH, kD, GF_RELU);
    gemm(hbuf, We2t + (size_t)e * kEH * kDO, be2 + (size_t)e * kDO,
         gates + e, kE, moe, nullptr, kN, kDO, kEH, e == 0 ? 0 : GF_ACCUM);
  }
  // ---- shared experts ----
  for (int s = 0; s < kES; ++s) {
    gemm(xb, Ws1t + (size_t)s * kD * kEH, bs1 + (size_t)s * kEH,
         nullptr, 0, nullptr, hbuf, kN, kEH, kD, GF_RELU);
    gemm(hbuf, Ws2t + (size_t)s * kEH * kDO, bs2 + (size_t)s * kDO,
         nullptr, 0, moe, nullptr, kN, kDO, kEH, GF_ACCUM);
  }

  // ---- cross-attention ----
  cvt(moe, objb, (size_t)kN * kDO);
  gemm(objb, Wqt, bq, nullptr, 0, nullptr, Qb, kN, kDO, kDO, 0);
  gemm(objb, Wkt, bk, nullptr, 0, nullptr, Kb, kN, kDO, kDO, 0);
  gemm(objb, Wvt, bv, nullptr, 0, nullptr, Vbt, kN, kDO, kDO, GF_TRANSB);
  attn_kernel<<<dim3(kS / 64, kBsz * kH), 128, 0, stream>>>(Qb, Kb, Vbt, ctxb);
  gemm(ctxb, Wot, bo, nullptr, 0, nullptr, relb, kN, kDO, kDO, 0);

  // ---- FFN ----
  gemm(relb, Wf1t, bf1, nullptr, 0, nullptr, h1b, kN, kFH, kDO, GF_RELU);
  gemm(h1b, Wf2t, bf2, nullptr, 0, out, nullptr, kN, kDO, kFH, 0);
}
